// ImageEncoder_1838246002910
// MI455X (gfx1250) — matrix operand order repaired and checked by reading, not compiled, not run on hardware
//
#include <hip/hip_runtime.h>

typedef __bf16 bf16;
typedef __attribute__((ext_vector_type(16))) __bf16 v16bf;
typedef __attribute__((ext_vector_type(8)))  float  v8f;

constexpr int CB  = 8;
constexpr int CS  = 2048;
constexpr int CP  = 768;
constexpr int CE  = 256;
constexpr int CH  = 4;
constexpr int CHD = 64;
constexpr int CM  = CB * CS;   // 16384 tokens

// ---------------------------------------------------------------------------
// WMMA fragment layouts used here (A/B: this code's contiguous k order, same for both):
// A (16x32, MxK): lane = half*16 + m ; elem i -> K = half*16 + i
// B (32x16, KxN): lane = half*16 + n ; elem i -> K = half*16 + i
// C/D (16x16)   : VGPR r -> row = r + 8*half, col = lane & 15
// ---------------------------------------------------------------------------
static __device__ inline v16bf load_a_frag(const bf16* __restrict__ base, int ld,
                                           int row0, int k0, int lane) {
  const int half = (lane >> 4) & 1;
  const int m    = lane & 15;
  const bf16* p  = base + (size_t)(row0 + m) * ld + k0 + half * 16;
  v16bf f;
#pragma unroll
  for (int i = 0; i < 8; ++i) f[i] = p[i];
#pragma unroll
  for (int i = 0; i < 8; ++i) f[8 + i] = p[8 + i];
  return f;
}

// B fragment from a row-major NxK source read transposed (K^T from K[S][HD])
static __device__ inline v16bf load_bT_frag(const bf16* __restrict__ base, int ld,
                                            int k0, int col0, int lane) {
  const int half = (lane >> 4) & 1;
  const int n    = lane & 15;
  const bf16* p  = base + (size_t)(col0 + n) * ld + k0 + half * 16;
  v16bf f;
#pragma unroll
  for (int i = 0; i < 16; ++i) f[i] = p[i];   // contiguous 32B per lane
  return f;
}

// B fragment from pre-swizzled storage: one contiguous 32B vector per lane.
// layout: [kt][nt][lane][i], i.e. ((kt*NT + nt)*32 + lane)*16 + i
static __device__ inline v16bf load_b_swz(const bf16* __restrict__ base, int NT,
                                          int kt, int nt, int lane) {
  const bf16* p = base + (((size_t)kt * NT + nt) * 32 + lane) * 16;
  return *(const v16bf*)p;
}

static __device__ inline v8f wmma_bf16(v16bf a, v16bf b, v8f c) {
  return __builtin_amdgcn_wmma_f32_16x16x32_bf16(false, a, false, b, (short)0, c,
                                                 false, false);
}

// ---------------------------------------------------------------------------
// fp32 -> bf16 elementwise conversion
// ---------------------------------------------------------------------------
__global__ void cvt_bf16_kernel(const float* __restrict__ src,
                                bf16* __restrict__ dst, int n) {
  for (int i = blockIdx.x * blockDim.x + threadIdx.x; i < n;
       i += gridDim.x * blockDim.x)
    dst[i] = (bf16)src[i];
}

// ---------------------------------------------------------------------------
// fp32 weight [K][N] (row-major) -> bf16 swizzled B-fragment layout
// dst[((kt*NT + nt)*32 + lane)*16 + i] = W[kt*32 + half*16 + i][nt*16 + idx]
// ---------------------------------------------------------------------------
__global__ void swizzle_weight_kernel(const float* __restrict__ W,
                                      bf16* __restrict__ dst, int K, int N) {
  const int NT    = N / 16;
  const int total = K * N;
  for (int f = blockIdx.x * blockDim.x + threadIdx.x; f < total;
       f += gridDim.x * blockDim.x) {
    const int i    = f & 15;
    const int lane = (f >> 4) & 31;
    const int tile = f >> 9;
    const int nt   = tile % NT;
    const int kt   = tile / NT;
    const int half = lane >> 4;
    const int idx  = lane & 15;
    dst[f] = (bf16)W[(size_t)(kt * 32 + half * 16 + i) * N + nt * 16 + idx];
  }
}

// ---------------------------------------------------------------------------
// Embedding GEMM [CM,CP]x[CP,CE] + bias + RoPE -> h (bf16)
// One wave per 32x64 output tile (2 A-frags x 4 B-frags -> 8 WMMA / k-step).
// ---------------------------------------------------------------------------
__global__ __launch_bounds__(32)
void embed_rope_kernel(const bf16* __restrict__ xb, const bf16* __restrict__ Wb,
                       const float* __restrict__ bias,
                       const int* __restrict__ lengths, bf16* __restrict__ hb) {
  const int lane = threadIdx.x & 31;
  const int ngrp = CE / 64;                 // 4
  const int mt   = blockIdx.x / ngrp;
  const int ng   = blockIdx.x % ngrp;
  const int row0 = mt * 32;
  const int col0 = ng * 64;
  const int NT   = CE / 16;

  v8f acc[2][4] = {};
  for (int k0 = 0; k0 < CP; k0 += 32) {
    v16bf a0 = load_a_frag(xb, CP, row0, k0, lane);
    v16bf a1 = load_a_frag(xb, CP, row0 + 16, k0, lane);
    if (k0 + 32 < CP)
      __builtin_prefetch(xb + (size_t)(row0 + (lane & 15)) * CP + k0 + 32, 0, 1);
#pragma unroll
    for (int t = 0; t < 4; ++t) {
      v16bf b   = load_b_swz(Wb, NT, k0 >> 5, (col0 >> 4) + t, lane);
      acc[0][t] = wmma_bf16(a0, b, acc[0][t]);
      acc[1][t] = wmma_bf16(a1, b, acc[1][t]);
    }
  }

  const int half  = (lane >> 4) & 1;
  const int idx   = lane & 15;
  const int bidx  = row0 / CS;              // whole tile in one batch (CS%32==0)
  const int len   = lengths[bidx];
  const int sbase = row0 % CS;

#pragma unroll
  for (int t = 0; t < 4; ++t) {
    const int   e     = col0 + t * 16 + idx;
    const int   d     = e & (CHD - 1);
    const float theta = __powf(10000.f, -(float)(d & ~1) / (float)CHD);
    const float bia   = bias[e];
#pragma unroll
    for (int sub = 0; sub < 2; ++sub)
#pragma unroll
      for (int r = 0; r < 8; ++r) {
        const int   row = sub * 16 + r + 8 * half;
        const float v   = acc[sub][t][r] + bia;
        const float pos = (float)(sbase + row - (len >> 1));
        float c, sn;
        __sincosf(pos * theta, &sn, &c);
        float partner = __shfl_xor(v, 1, 32);  // pair column lives in lane^1
        float out = (d & 1) ? (v * c + partner * sn) : (v * c - partner * sn);
        hb[(size_t)(row0 + row) * CE + e] = (bf16)out;
      }
  }
}

// ---------------------------------------------------------------------------
// Fused QKV GEMM: h[CM,CE] x {Wq,Wk,Wv}[CE,CE] + bias (16-row tiles; one
// A-frag already feeds 12 WMMAs). q,k -> [B,H,S,HD]; v -> swizzled B layout
// per (b,h): ((jt*4 + nt)*32 + lane)*16 + i  with jt = s/32.
// ---------------------------------------------------------------------------
__global__ __launch_bounds__(32)
void qkv_kernel(const bf16* __restrict__ hb, const bf16* __restrict__ Wq,
                const bf16* __restrict__ Wk, const bf16* __restrict__ Wv,
                const float* __restrict__ bq, const float* __restrict__ bk,
                const float* __restrict__ bv, bf16* __restrict__ qo,
                bf16* __restrict__ ko, bf16* __restrict__ vswz) {
  const int lane = threadIdx.x & 31;
  const int ngrp = CE / 64;
  const int mt   = blockIdx.x / ngrp;
  const int ng   = blockIdx.x % ngrp;
  const int row0 = mt * 16;
  const int col0 = ng * 64;
  const int NT   = CE / 16;

  v8f aq[4] = {}, ak[4] = {}, av[4] = {};
  for (int k0 = 0; k0 < CE; k0 += 32) {
    v16bf a = load_a_frag(hb, CE, row0, k0, lane);
    if (k0 + 32 < CE)
      __builtin_prefetch(hb + (size_t)(row0 + (lane & 15)) * CE + k0 + 32, 0, 1);
    const int kt = k0 >> 5;
#pragma unroll
    for (int t = 0; t < 4; ++t) {
      const int nt = (col0 >> 4) + t;
      aq[t] = wmma_bf16(a, load_b_swz(Wq, NT, kt, nt, lane), aq[t]);
      ak[t] = wmma_bf16(a, load_b_swz(Wk, NT, kt, nt, lane), ak[t]);
      av[t] = wmma_bf16(a, load_b_swz(Wv, NT, kt, nt, lane), av[t]);
    }
  }

  const int half = (lane >> 4) & 1;
  const int idx  = lane & 15;
  const int head = col0 >> 6;               // == ng
#pragma unroll
  for (int t = 0; t < 4; ++t) {
    const int e    = col0 + t * 16 + idx;
    const int d    = e & (CHD - 1);
    const float biq = bq[e], bik = bk[e], biv = bv[e];
#pragma unroll
    for (int r = 0; r < 8; ++r) {
      const int token = row0 + r + 8 * half;
      const int b     = token / CS;
      const int s     = token % CS;
      const size_t bh = (size_t)b * CH + head;
      const size_t off = (bh * CS + s) * CHD + d;
      qo[off] = (bf16)(aq[t][r] + biq);
      ko[off] = (bf16)(ak[t][r] + bik);
      // V in swizzled B-fragment layout for the P*V WMMA
      const int jt   = s >> 5;
      const int rel  = s & 31;
      const size_t voff =
          bh * (CS * CHD) +
          (((size_t)(jt * 4 + t) * 32) + ((rel >> 4) * 16 + idx)) * 16 +
          (rel & 15);
      vswz[voff] = (bf16)(av[t][r] + biv);
    }
  }
}

// ---------------------------------------------------------------------------
// Flash-style attention. One wave per (b, h, 32-query tile): every K^T / V
// fragment is reused by two Q tiles -> 16 WMMAs per 32-key step vs 8 loads.
// P transposed C-layout -> A-layout through LDS; V read pre-swizzled.
// ---------------------------------------------------------------------------
__global__ __launch_bounds__(32)
void attn_kernel(const bf16* __restrict__ qb, const bf16* __restrict__ kb,
                 const bf16* __restrict__ vswz, const int* __restrict__ lengths,
                 bf16* __restrict__ ob) {
  __shared__ bf16 Pt[32 * 32];

  const int lane = threadIdx.x & 31;
  const int nqt  = CS / 32;
  const int qt   = blockIdx.x % nqt;
  const int bh   = blockIdx.x / nqt;
  const int b    = bh / CH;
  const int h    = bh % CH;
  const size_t base = ((size_t)b * CH + h) * CS * CHD;
  const bf16* Q  = qb + base;
  const bf16* K  = kb + base;
  const bf16* Vs = vswz + base;
  const int len  = lengths[b];
  const int half = (lane >> 4) & 1;
  const int idx  = lane & 15;
  const int q0   = qt * 32;

  v16bf qf[2][2];
#pragma unroll
  for (int s = 0; s < 2; ++s) {
    qf[s][0] = load_a_frag(Q, CHD, q0 + s * 16, 0, lane);
    qf[s][1] = load_a_frag(Q, CHD, q0 + s * 16, 32, lane);
  }

  float m[2][8], l[2][8];
#pragma unroll
  for (int s = 0; s < 2; ++s)
#pragma unroll
    for (int r = 0; r < 8; ++r) { m[s][r] = -1e30f; l[s][r] = 0.f; }
  v8f o[2][4] = {};

  for (int j0 = 0; j0 < CS; j0 += 32) {
    if (j0 >= len) break;   // uniform per wave: fully-masked tail contributes 0

    // logits: Q(32x64) * K^T(64x32) as four 16x16 C-tiles; K frags shared
    const v16bf kb0a = load_bT_frag(K, CHD, 0,  j0,      lane);
    const v16bf kb0b = load_bT_frag(K, CHD, 32, j0,      lane);
    const v16bf kb1a = load_bT_frag(K, CHD, 0,  j0 + 16, lane);
    const v16bf kb1b = load_bT_frag(K, CHD, 32, j0 + 16, lane);

    v8f sc[2][2] = {};
#pragma unroll
    for (int s = 0; s < 2; ++s) {
      sc[s][0] = wmma_bf16(qf[s][0], kb0a, sc[s][0]);
      sc[s][0] = wmma_bf16(qf[s][1], kb0b, sc[s][0]);
      sc[s][1] = wmma_bf16(qf[s][0], kb1a, sc[s][1]);
      sc[s][1] = wmma_bf16(qf[s][1], kb1b, sc[s][1]);
    }

    const bool mk0 = (j0 + idx) >= len;
    const bool mk1 = (j0 + 16 + idx) >= len;

#pragma unroll
    for (int s = 0; s < 2; ++s) {
      float t0[8], t1[8];
#pragma unroll
      for (int r = 0; r < 8; ++r) {
        t0[r] = mk0 ? -1e30f : sc[s][0][r] * 0.125f;   // 1/sqrt(64)
        t1[r] = mk1 ? -1e30f : sc[s][1][r] * 0.125f;
      }
      float alpha[8];
#pragma unroll
      for (int r = 0; r < 8; ++r) {
        float v = fmaxf(t0[r], t1[r]);
#pragma unroll
        for (int off = 8; off >= 1; off >>= 1)
          v = fmaxf(v, __shfl_xor(v, off, 32));
        const float mn = fmaxf(m[s][r], v);
        alpha[r]       = __expf(m[s][r] - mn);
        m[s][r]        = mn;
      }
#pragma unroll
      for (int r = 0; r < 8; ++r) {
        t0[r]    = __expf(t0[r] - m[s][r]);
        t1[r]    = __expf(t1[r] - m[s][r]);
        float sm = t0[r] + t1[r];
#pragma unroll
        for (int off = 8; off >= 1; off >>= 1) sm += __shfl_xor(sm, off, 32);
        l[s][r] = l[s][r] * alpha[r] + sm;
      }
#pragma unroll
      for (int n = 0; n < 4; ++n)
#pragma unroll
        for (int r = 0; r < 8; ++r) o[s][n][r] *= alpha[r];

      // stage P (C-layout) into LDS, rows [s*16, s*16+16)
#pragma unroll
      for (int r = 0; r < 8; ++r) {
        const int row = s * 16 + r + 8 * half;
        Pt[row * 32 + idx]      = (bf16)t0[r];
        Pt[row * 32 + 16 + idx] = (bf16)t1[r];
      }
    }
    asm volatile("s_wait_dscnt 0" ::: "memory");
    __syncthreads();  // single-wave WG: barrier is a NOP, fence orders LDS

    const int jt = j0 >> 5;
#pragma unroll
    for (int s = 0; s < 2; ++s) {
      const v16bf pa = load_a_frag(Pt, 32, s * 16, 0, lane);
#pragma unroll
      for (int n = 0; n < 4; ++n) {
        v16bf bv = load_b_swz(Vs, 1, jt * 4 + n, 0, lane);  // contiguous frag
        o[s][n]  = wmma_bf16(pa, bv, o[s][n]);
      }
    }
    asm volatile("s_wait_dscnt 0" ::: "memory");
    __syncthreads();
  }

  // epilogue: normalize and store in [B,S,E] bf16 (len >= S/2 so l > 0)
#pragma unroll
  for (int s = 0; s < 2; ++s)
#pragma unroll
    for (int n = 0; n < 4; ++n)
#pragma unroll
      for (int r = 0; r < 8; ++r) {
        const int row = s * 16 + r + 8 * half;
        const float val = o[s][n][r] / l[s][r];
        ob[((size_t)b * CS + q0 + row) * CE + h * CHD + n * 16 + idx] =
            (bf16)val;
      }
}

// ---------------------------------------------------------------------------
// Output projection: o[CM,CE] x Wo[CE,CE] + bo -> d_out (fp32), 32-row tiles
// ---------------------------------------------------------------------------
__global__ __launch_bounds__(32)
void proj_kernel(const bf16* __restrict__ ob, const bf16* __restrict__ Wo,
                 const float* __restrict__ bo, float* __restrict__ out) {
  const int lane = threadIdx.x & 31;
  const int ngrp = CE / 64;
  const int mt   = blockIdx.x / ngrp;
  const int ng   = blockIdx.x % ngrp;
  const int row0 = mt * 32;
  const int col0 = ng * 64;
  const int NT   = CE / 16;

  v8f acc[2][4] = {};
  for (int k0 = 0; k0 < CE; k0 += 32) {
    v16bf a0 = load_a_frag(ob, CE, row0, k0, lane);
    v16bf a1 = load_a_frag(ob, CE, row0 + 16, k0, lane);
    if (k0 + 32 < CE)
      __builtin_prefetch(ob + (size_t)(row0 + (lane & 15)) * CE + k0 + 32, 0, 1);
#pragma unroll
    for (int t = 0; t < 4; ++t) {
      v16bf b   = load_b_swz(Wo, NT, k0 >> 5, (col0 >> 4) + t, lane);
      acc[0][t] = wmma_bf16(a0, b, acc[0][t]);
      acc[1][t] = wmma_bf16(a1, b, acc[1][t]);
    }
  }

  const int half = (lane >> 4) & 1;
  const int idx  = lane & 15;
#pragma unroll
  for (int t = 0; t < 4; ++t) {
    const int e    = col0 + t * 16 + idx;
    const float bi = bo[e];
#pragma unroll
    for (int sub = 0; sub < 2; ++sub)
#pragma unroll
      for (int r = 0; r < 8; ++r)
        out[(size_t)(row0 + sub * 16 + r + 8 * half) * CE + e] =
            acc[sub][t][r] + bi;
  }
}

// ---------------------------------------------------------------------------
extern "C" void kernel_launch(void* const* d_in, const int* in_sizes, int n_in,
                              void* d_out, int out_size, void* d_ws,
                              size_t ws_size, hipStream_t stream) {
  const float* x     = (const float*)d_in[0];
  const int*   lens  = (const int*)d_in[1];
  const float* W_emb = (const float*)d_in[2];
  const float* b_emb = (const float*)d_in[3];
  const float* Wq    = (const float*)d_in[4];
  const float* bq    = (const float*)d_in[5];
  const float* Wk    = (const float*)d_in[6];
  const float* bk    = (const float*)d_in[7];
  const float* Wv    = (const float*)d_in[8];
  const float* bv    = (const float*)d_in[9];
  const float* Wo    = (const float*)d_in[10];
  const float* bo    = (const float*)d_in[11];
  float* out = (float*)d_out;

  // workspace layout (bf16 buffers), 256B-aligned offsets
  char* ws = (char*)d_ws;
  size_t off = 0;
  auto alloc = [&](size_t bytes) {
    char* p = ws + off;
    off += (bytes + 255) & ~(size_t)255;
    return p;
  };
  bf16* xb  = (bf16*)alloc((size_t)CM * CP * 2);
  bf16* hb  = (bf16*)alloc((size_t)CM * CE * 2);
  bf16* qb  = (bf16*)alloc((size_t)CM * CE * 2);
  bf16* kb  = (bf16*)alloc((size_t)CM * CE * 2);
  bf16* vb  = (bf16*)alloc((size_t)CM * CE * 2);   // swizzled V
  bf16* ob  = (bf16*)alloc((size_t)CM * CE * 2);
  bf16* Web = (bf16*)alloc((size_t)CP * CE * 2);   // swizzled weights
  bf16* Wqb = (bf16*)alloc((size_t)CE * CE * 2);
  bf16* Wkb = (bf16*)alloc((size_t)CE * CE * 2);
  bf16* Wvb = (bf16*)alloc((size_t)CE * CE * 2);
  bf16* Wob = (bf16*)alloc((size_t)CE * CE * 2);
  (void)ws_size;

  {
    int n = CM * CP;
    int blocks = (n + 255) / 256;
    if (blocks > 4096) blocks = 4096;
    cvt_bf16_kernel<<<blocks, 256, 0, stream>>>(x, xb, n);
  }
  auto swz = [&](const float* s, bf16* d, int K, int N) {
    int n = K * N;
    int blocks = (n + 255) / 256;
    if (blocks > 2048) blocks = 2048;
    swizzle_weight_kernel<<<blocks, 256, 0, stream>>>(s, d, K, N);
  };
  swz(W_emb, Web, CP, CE);
  swz(Wq, Wqb, CE, CE);
  swz(Wk, Wkb, CE, CE);
  swz(Wv, Wvb, CE, CE);
  swz(Wo, Wob, CE, CE);

  embed_rope_kernel<<<(CM / 32) * (CE / 64), 32, 0, stream>>>(xb, Web, b_emb,
                                                              lens, hb);
  qkv_kernel<<<(CM / 16) * (CE / 64), 32, 0, stream>>>(hb, Wqb, Wkb, Wvb, bq,
                                                       bk, bv, qb, kb, vb);
  attn_kernel<<<CB * CH * (CS / 32), 32, 0, stream>>>(qb, kb, vb, lens, ob);
  proj_kernel<<<(CM / 32) * (CE / 64), 32, 0, stream>>>(ob, Wob, bo, out);
  (void)in_sizes; (void)n_in; (void)out_size;
}
